// SelfAttention_39496519254679
// MI455X (gfx1250) — matrix operand order repaired and checked by reading, not compiled, not run on hardware
//
#include <hip/hip_runtime.h>
#include <math.h>

// ---------------- types ----------------
typedef _Float16 half_t;
typedef __attribute__((ext_vector_type(16))) _Float16    v16h;
typedef __attribute__((ext_vector_type(8)))  float       v8f;
typedef __attribute__((ext_vector_type(4)))  unsigned int u32x4;

#define B_    4
#define S_    2048
#define DIN   1024
#define DOUT  1024
#define NEGV  (-1000000000.0f)

union FragU { u32x4 u[2]; v16h h; };

// A fragment (16x32 f16, MxK). lane<16: K = kb+0..7 | kb+16..23 ; lane>=16: +8.
// base = row pointer + kb (halves). hl = lane>>4.
static __device__ __forceinline__ v16h load_frag_a(const half_t* base, int hl) {
    FragU r;
    const char* p = (const char*)base + hl * 16;
    r.u[0] = *(const u32x4*)(p);
    r.u[1] = *(const u32x4*)(p + 32);
    return r.h;
}

// B fragment (32x16 f16, KxN). lane holds column n=lane&15; lane<16: K=kb+0..7 | kb+16..23,
// lane>=16: +8. base = source row (column of B) pointer + kb (halves).
static __device__ __forceinline__ v16h load_frag_b(const half_t* base, int hl) {
    FragU r;
    const char* p = (const char*)base + hl * 16;
    r.u[0] = *(const u32x4*)(p);
    r.u[1] = *(const u32x4*)(p + 32);
    return r.h;
}

static __device__ __forceinline__ v8f wmma16(v16h a, v16h b, v8f c) {
    return __builtin_amdgcn_wmma_f32_16x16x32_f16(
        /*neg_a=*/false, a, /*neg_b=*/false, b,
        /*c_mod=*/(short)0, c, /*reuse_a=*/false, /*reuse_b=*/false);
}

// async global(16B/lane) -> LDS, tracked by ASYNCcnt
static __device__ __forceinline__ void async_g2l_b128(unsigned lds_off, const void* gaddr) {
    asm volatile("global_load_async_to_lds_b128 %0, %1, off"
                 :: "v"(lds_off), "v"(gaddr) : "memory");
}

// ---------------- 1) fp32 -> fp16 convert ----------------
__global__ void __launch_bounds__(256)
cvt_f32_f16_kernel(const float* __restrict__ src, half_t* __restrict__ dst, int n) {
    int i = blockIdx.x * blockDim.x + threadIdx.x;
    int stride = gridDim.x * blockDim.x;
    for (; i < n; i += stride) dst[i] = (half_t)src[i];
}

// ---------------- 2) QKV projection GEMM (LDS double-buffered async) --------
// Y[r, o] = (sum_d X[r,d] * W[o,d] + bias[o]) * scale ; M=8192, N=K=1024
// Block: 256 thr = 8 waves -> 64x128 output tile; wave = 32x32 (2x2 WMMA tiles)
// K staged in chunks of 64 halves; chunk c+1 streams into LDS buffer ^1 via
// global_load_async_to_lds_b128 while chunk c is consumed (s_wait_asynccnt 6
// releases only the oldest chunk: async loads complete in order).
// tr==0: out row-major [M, DOUT] ; tr==1: per-batch transposed Vt[(b*DOUT+o)*S+s]
#define SA      72      // padded LDS row stride in halves (144B = 36 banks)
#define NCHUNK  16      // 1024 / 64
__global__ void __launch_bounds__(256)
qkv_gemm_kernel(const half_t* __restrict__ X, const half_t* __restrict__ W,
                const float* __restrict__ bias, half_t* __restrict__ out,
                float scale, int tr) {
    __shared__ half_t As[2][64  * SA];   // 2 x  9216 B
    __shared__ half_t Bs[2][128 * SA];   // 2 x 18432 B   (total 55296 B)

    const int t    = threadIdx.x;
    const int wave = t >> 5;
    const int lane = t & 31;
    const int l15  = lane & 15, hl = lane >> 4;
    const int wm   = wave & 1;        // 0..1  (M direction, 32 rows each)
    const int wn   = wave >> 1;       // 0..3  (N direction, 32 cols each)
    const int bm   = blockIdx.x;      // 0..127 (64 rows each)
    const int bn   = blockIdx.y;      // 0..7   (128 cols each)

    // per-thread staging coordinates (6 x 16B transfers per chunk)
    const int rA0 = t >> 3,          sA0 = t & 7;           // A transfer 0
    const int rA1 = (256 + t) >> 3,  sA1 = t & 7;           // A transfer 1
    const half_t* xr0 = X + (size_t)(bm * 64 + rA0) * DIN + sA0 * 8;
    const half_t* xr1 = X + (size_t)(bm * 64 + rA1) * DIN + sA1 * 8;

    v8f acc00 = {}, acc01 = {}, acc10 = {}, acc11 = {};

    // ---- stage chunk 0 into buffer 0 ----
    {
        const unsigned aB = (unsigned)(unsigned long long)(void*)As[0];
        const unsigned bB = (unsigned)(unsigned long long)(void*)Bs[0];
        async_g2l_b128(aB + (unsigned)((rA0 * SA + sA0 * 8) * 2), xr0);
        async_g2l_b128(aB + (unsigned)((rA1 * SA + sA1 * 8) * 2), xr1);
        #pragma unroll
        for (int i = 0; i < 4; ++i) {
            const int id = i * 256 + t;
            const int row = id >> 3, seg = id & 7;
            const half_t* src = W + (size_t)(bn * 128 + row) * DIN + seg * 8;
            async_g2l_b128(bB + (unsigned)((row * SA + seg * 8) * 2), src);
        }
    }

    for (int c = 0; c < NCHUNK; ++c) {
        const int buf = c & 1;
        const int kb1 = (c + 1) * 64;
        if (c + 1 < NCHUNK) {
            // ---- stage chunk c+1 into the other buffer (overlaps compute) ----
            const unsigned aB = (unsigned)(unsigned long long)(void*)As[buf ^ 1];
            const unsigned bB = (unsigned)(unsigned long long)(void*)Bs[buf ^ 1];
            async_g2l_b128(aB + (unsigned)((rA0 * SA + sA0 * 8) * 2), xr0 + kb1);
            async_g2l_b128(aB + (unsigned)((rA1 * SA + sA1 * 8) * 2), xr1 + kb1);
            #pragma unroll
            for (int i = 0; i < 4; ++i) {
                const int id = i * 256 + t;
                const int row = id >> 3, seg = id & 7;
                const half_t* src = W + (size_t)(bn * 128 + row) * DIN + kb1 + seg * 8;
                async_g2l_b128(bB + (unsigned)((row * SA + seg * 8) * 2), src);
            }
            // oldest 6 (chunk c) complete; chunk c+1 may still be in flight
            asm volatile("s_wait_asynccnt 0x6" ::: "memory");
        } else {
            asm volatile("s_wait_asynccnt 0x0" ::: "memory");
        }
        __syncthreads();

        // ---- compute chunk c: 2 K-steps x (2x2) WMMA ----
        #pragma unroll
        for (int ks = 0; ks < 64; ks += 32) {
            v16h a0 = load_frag_a(&As[buf][(wm * 32 +  0 + l15) * SA + ks], hl);
            v16h a1 = load_frag_a(&As[buf][(wm * 32 + 16 + l15) * SA + ks], hl);
            v16h b0 = load_frag_b(&Bs[buf][(wn * 32 +  0 + l15) * SA + ks], hl);
            v16h b1 = load_frag_b(&Bs[buf][(wn * 32 + 16 + l15) * SA + ks], hl);
            acc00 = wmma16(a0, b0, acc00);
            acc01 = wmma16(a0, b1, acc01);
            acc10 = wmma16(a1, b0, acc10);
            acc11 = wmma16(a1, b1, acc11);
        }
        __syncthreads();   // all reads of buf done before it is restaged
    }

    // ---- epilogue ----
    const float bn0 = bias[bn * 128 + wn * 32 +  0 + l15];
    const float bn1 = bias[bn * 128 + wn * 32 + 16 + l15];
    #pragma unroll
    for (int a = 0; a < 2; ++a) {
        const v8f* accA = a ? &acc10 : &acc00;  // [a][b2=0]
        const v8f* accB = a ? &acc11 : &acc01;  // [a][b2=1]
        #pragma unroll
        for (int v = 0; v < 8; ++v) {
            const int m = v + 8 * hl;
            const int r = bm * 64 + wm * 32 + a * 16 + m;
            const float v0 = ((*accA)[v] + bn0) * scale;
            const float v1 = ((*accB)[v] + bn1) * scale;
            const int o0 = bn * 128 + wn * 32 + l15;
            const int o1 = o0 + 16;
            if (!tr) {
                out[(size_t)r * DOUT + o0] = (half_t)v0;
                out[(size_t)r * DOUT + o1] = (half_t)v1;
            } else {
                const int b = r >> 11, s = r & (S_ - 1);
                out[((size_t)(b * DOUT + o0)) * S_ + s] = (half_t)v0;
                out[((size_t)(b * DOUT + o1)) * S_ + s] = (half_t)v1;
            }
        }
    }
}

// ---------------- 3) scores = Q K^T (+causal/len masks, additive like ref) ----
// Wave computes a 32x32 score tile (2x2 register blocking).
__global__ void __launch_bounds__(256)
scores_kernel(const half_t* __restrict__ Q, const half_t* __restrict__ Km,
              const int* __restrict__ lengths, float* __restrict__ Sc) {
    const int b = blockIdx.y;
    const int wave = threadIdx.x >> 5;
    const int lane = threadIdx.x & 31;
    const int tile = blockIdx.x * 8 + wave;      // 4096 tiles / batch
    const int it = tile >> 6, jt = tile & 63;    // 0..63 each
    const int l15 = lane & 15, hl = lane >> 4;
    const int i0 = it * 32, j0 = jt * 32;
    const int len = lengths[b];
    float* outp = Sc + ((size_t)b * S_ + i0) * S_ + j0;

    if (j0 > i0 + 31) {  // fully causal-masked 32x32 tile: skip GEMM
        #pragma unroll
        for (int a = 0; a < 2; ++a)
        #pragma unroll
        for (int b2 = 0; b2 < 2; ++b2)
        #pragma unroll
        for (int v = 0; v < 8; ++v) {
            const int m = a * 16 + v + 8 * hl;
            const int i = i0 + m, j = j0 + b2 * 16 + l15;
            float msk = NEGV;                              // causal
            if (i >= len || j >= len) msk += NEGV;         // length
            outp[(size_t)m * S_ + b2 * 16 + l15] = msk;
        }
        return;
    }

    const half_t* ar0 = Q  + ((size_t)(b * S_ + i0 +  0 + l15)) * DOUT;
    const half_t* ar1 = Q  + ((size_t)(b * S_ + i0 + 16 + l15)) * DOUT;
    const half_t* br0 = Km + ((size_t)(b * S_ + j0 +  0 + l15)) * DOUT;
    const half_t* br1 = Km + ((size_t)(b * S_ + j0 + 16 + l15)) * DOUT;
    v8f acc00 = {}, acc01 = {}, acc10 = {}, acc11 = {};
    #pragma unroll 2
    for (int kb = 0; kb < DOUT; kb += 32) {
        v16h a0 = load_frag_a(ar0 + kb, hl);
        v16h a1 = load_frag_a(ar1 + kb, hl);
        v16h b0 = load_frag_b(br0 + kb, hl);
        v16h b1 = load_frag_b(br1 + kb, hl);
        acc00 = wmma16(a0, b0, acc00);
        acc01 = wmma16(a0, b1, acc01);
        acc10 = wmma16(a1, b0, acc10);
        acc11 = wmma16(a1, b1, acc11);
    }
    #pragma unroll
    for (int a = 0; a < 2; ++a) {
        const v8f* accA = a ? &acc10 : &acc00;
        const v8f* accB = a ? &acc11 : &acc01;
        #pragma unroll
        for (int v = 0; v < 8; ++v) {
            const int m = a * 16 + v + 8 * hl;
            const int i = i0 + m;
            const int jA = j0 + l15, jB = jA + 16;
            float s0 = (*accA)[v], s1 = (*accB)[v];       // 1/sqrt(D) folded in Q
            if (jA > i) s0 += NEGV;
            if (jB > i) s1 += NEGV;
            if (i >= len || jA >= len) s0 += NEGV;
            if (i >= len || jB >= len) s1 += NEGV;
            outp[(size_t)m * S_ + l15]      = s0;
            outp[(size_t)m * S_ + 16 + l15] = s1;
        }
    }
}

// ---------------- 4) row softmax (fp32) -> fp16 P ----------------
__global__ void __launch_bounds__(256)
softmax_kernel(const float* __restrict__ Sc, half_t* __restrict__ P) {
    const int row = blockIdx.x;                  // 0..B*S-1
    const float* src = Sc + (size_t)row * S_;
    half_t*      dst = P  + (size_t)row * S_;
    const int t = threadIdx.x;                   // 256 threads, 8 cols each
    const int w = t >> 5, ln = t & 31;
    __shared__ float red[8];

    float vals[8];
    float mx = -INFINITY;
    #pragma unroll
    for (int e = 0; e < 8; ++e) { vals[e] = src[t + e * 256]; mx = fmaxf(mx, vals[e]); }
    #pragma unroll
    for (int msk = 16; msk >= 1; msk >>= 1) mx = fmaxf(mx, __shfl_xor(mx, msk, 32));
    if (ln == 0) red[w] = mx;
    __syncthreads();
    if (t == 0) { float v = red[0]; for (int i = 1; i < 8; ++i) v = fmaxf(v, red[i]); red[0] = v; }
    __syncthreads();
    mx = red[0];
    __syncthreads();

    float sum = 0.0f;
    #pragma unroll
    for (int e = 0; e < 8; ++e) { vals[e] = __expf(vals[e] - mx); sum += vals[e]; }
    #pragma unroll
    for (int msk = 16; msk >= 1; msk >>= 1) sum += __shfl_xor(sum, msk, 32);
    if (ln == 0) red[w] = sum;
    __syncthreads();
    if (t == 0) { float v = 0.0f; for (int i = 0; i < 8; ++i) v += red[i]; red[0] = v; }
    __syncthreads();
    const float inv = 1.0f / red[0];
    #pragma unroll
    for (int e = 0; e < 8; ++e) dst[t + e * 256] = (half_t)(vals[e] * inv);
}

// ---------------- 5) O = P * V  (V pre-transposed) -- 32x32 per wave ---------
__global__ void __launch_bounds__(256)
out_gemm_kernel(const half_t* __restrict__ P, const half_t* __restrict__ Vt,
                float* __restrict__ O) {
    const int b = blockIdx.y;
    const int wave = threadIdx.x >> 5;
    const int lane = threadIdx.x & 31;
    const int tile = blockIdx.x * 8 + wave;      // 2048 tiles / batch
    const int it = tile >> 5;                    // 0..63
    const int nt = tile & 31;                    // 0..31
    const int l15 = lane & 15, hl = lane >> 4;
    const int i0 = it * 32, n0 = nt * 32;

    const half_t* ar0 = P  + ((size_t)(b * S_   + i0 +  0 + l15)) * S_;
    const half_t* ar1 = P  + ((size_t)(b * S_   + i0 + 16 + l15)) * S_;
    const half_t* br0 = Vt + ((size_t)(b * DOUT + n0 +  0 + l15)) * S_;
    const half_t* br1 = Vt + ((size_t)(b * DOUT + n0 + 16 + l15)) * S_;
    v8f acc00 = {}, acc01 = {}, acc10 = {}, acc11 = {};
    const int kend = i0 + 32;                    // causal: P[i][j>i] == 0 exactly
    for (int kb = 0; kb < kend; kb += 32) {
        v16h a0 = load_frag_a(ar0 + kb, hl);
        v16h a1 = load_frag_a(ar1 + kb, hl);
        v16h b0 = load_frag_b(br0 + kb, hl);
        v16h b1 = load_frag_b(br1 + kb, hl);
        acc00 = wmma16(a0, b0, acc00);
        acc01 = wmma16(a0, b1, acc01);
        acc10 = wmma16(a1, b0, acc10);
        acc11 = wmma16(a1, b1, acc11);
    }
    float* orow = O + ((size_t)(b * S_ + i0)) * DOUT + n0;
    #pragma unroll
    for (int a = 0; a < 2; ++a) {
        const v8f* accA = a ? &acc10 : &acc00;
        const v8f* accB = a ? &acc11 : &acc01;
        #pragma unroll
        for (int v = 0; v < 8; ++v) {
            const int m = a * 16 + v + 8 * hl;
            orow[(size_t)m * DOUT + l15]      = (*accA)[v];
            orow[(size_t)m * DOUT + 16 + l15] = (*accB)[v];
        }
    }
}

// ---------------- launch ----------------
extern "C" void kernel_launch(void* const* d_in, const int* in_sizes, int n_in,
                              void* d_out, int out_size, void* d_ws, size_t ws_size,
                              hipStream_t stream) {
    const float* x       = (const float*)d_in[0];
    const float* Wq      = (const float*)d_in[1];
    const float* bq      = (const float*)d_in[2];
    const float* Wk      = (const float*)d_in[3];
    const float* bk      = (const float*)d_in[4];
    const float* Wv      = (const float*)d_in[5];
    const float* bv      = (const float*)d_in[6];
    const int*   lengths = (const int*)d_in[7];
    float* out = (float*)d_out;

    const size_t M = (size_t)B_ * S_;            // 8192
    half_t* xb  = (half_t*)d_ws;                 // [M, DIN]
    half_t* wqb = xb  + M * DIN;                 // [DOUT, DIN]
    half_t* wkb = wqb + (size_t)DOUT * DIN;
    half_t* wvb = wkb + (size_t)DOUT * DIN;
    half_t* Qb  = wvb + (size_t)DOUT * DIN;      // [M, DOUT]
    half_t* Kb  = Qb  + M * DOUT;                // [M, DOUT]
    half_t* Vt  = Kb  + M * DOUT;                // [B*DOUT, S]
    half_t* Pb  = Vt  + M * DOUT;                // [M, S]
    float*  Sc  = (float*)(Pb + M * S_);         // [B, S, S]

    // 1) convert inputs to fp16
    cvt_f32_f16_kernel<<<2048, 256, 0, stream>>>(x,  xb,  (int)(M * DIN));
    cvt_f32_f16_kernel<<<1024, 256, 0, stream>>>(Wq, wqb, DOUT * DIN);
    cvt_f32_f16_kernel<<<1024, 256, 0, stream>>>(Wk, wkb, DOUT * DIN);
    cvt_f32_f16_kernel<<<1024, 256, 0, stream>>>(Wv, wvb, DOUT * DIN);

    // 2) QKV projections (1/sqrt(DOUT)=1/32 folded into Q; V stored transposed)
    qkv_gemm_kernel<<<dim3(128, 8), 256, 0, stream>>>(xb, wqb, bq, Qb, 1.0f / 32.0f, 0);
    qkv_gemm_kernel<<<dim3(128, 8), 256, 0, stream>>>(xb, wkb, bk, Kb, 1.0f, 0);
    qkv_gemm_kernel<<<dim3(128, 8), 256, 0, stream>>>(xb, wvb, bv, Vt, 1.0f, 1);

    // 3) masked scores
    scores_kernel<<<dim3(512, B_), 256, 0, stream>>>(Qb, Kb, lengths, Sc);

    // 4) softmax
    softmax_kernel<<<8192, 256, 0, stream>>>(Sc, Pb);

    // 5) O = P * V
    out_gemm_kernel<<<dim3(256, B_), 256, 0, stream>>>(Pb, Vt, out);
}
